// RandomMaskSubgraphs_71296457114203
// MI455X (gfx1250) — matrix operand order repaired and checked by reading, not compiled, not run on hardware
//
#include <hip/hip_runtime.h>
#include <hip/hip_bf16.h>
#include <stdint.h>

typedef __attribute__((ext_vector_type(16))) _Float16 v16h;
typedef __attribute__((ext_vector_type(8)))  float    v8f;

#define NNODES 150000
#define RS_THREADS 64
#define RS_CPT 64
#define RS_TILE (RS_THREADS * RS_CPT)   // 4096 keys per block
#define SCAN_THREADS 1024

// ---------------- Threefry-2x32 (JAX PRNG family) ----------------
__device__ __forceinline__ void tf2x32(uint32_t k0, uint32_t k1, uint32_t c0, uint32_t c1,
                                       uint32_t& o0, uint32_t& o1) {
  uint32_t ks2 = 0x1BD11BDAu ^ k0 ^ k1;
  uint32_t x0 = c0 + k0, x1 = c1 + k1;
#define TFR(r) { x0 += x1; x1 = (x1 << (r)) | (x1 >> (32 - (r))); x1 ^= x0; }
  TFR(13) TFR(15) TFR(26) TFR(6)
  x0 += k1;  x1 += ks2 + 1u;
  TFR(17) TFR(29) TFR(16) TFR(24)
  x0 += ks2; x1 += k0 + 2u;
  TFR(13) TFR(15) TFR(26) TFR(6)
  x0 += k0;  x1 += k1 + 3u;
  TFR(17) TFR(29) TFR(16) TFR(24)
  x0 += k1;  x1 += ks2 + 4u;
  TFR(13) TFR(15) TFR(26) TFR(6)
  x0 += ks2; x1 += k0 + 5u;
#undef TFR
  o0 = x0; o1 = x1;
}

// ---------------- BFS mask stages ----------------
__global__ void k_scatter_seeds(const int* __restrict__ seeds, int ns,
                                unsigned* __restrict__ seedA, unsigned* __restrict__ mask) {
  int i = blockIdx.x * blockDim.x + threadIdx.x;
  if (i < ns) { int s = seeds[i]; seedA[s] = 1u; mask[s] = 1u; }
}

__global__ void k_bfs0(const int* __restrict__ rows, const int* __restrict__ cols, int E,
                       const unsigned* __restrict__ seedA, unsigned* __restrict__ removed,
                       unsigned* __restrict__ seedB, unsigned* __restrict__ mask) {
  int e = blockIdx.x * blockDim.x + threadIdx.x;
  if (e >= E) return;
  int r = rows[e], c = cols[e];
  if (seedA[r] | seedA[c]) {
    removed[e] = 1u;
    seedB[r] = 1u; seedB[c] = 1u;
    mask[r]  = 1u; mask[c]  = 1u;
  }
}

__global__ void k_bfs1(const int* __restrict__ rows, const int* __restrict__ cols, int E,
                       const unsigned* __restrict__ seedB, unsigned* __restrict__ removed) {
  int e = blockIdx.x * blockDim.x + threadIdx.x;
  if (e >= E) return;
  int r = rows[e], c = cols[e];
  if (seedB[r] | seedB[c]) removed[e] = 1u;   // removed |= idct
}

// ---------------- Encoder adjacency ----------------
__global__ void k_degree(const int* __restrict__ rows, int E,
                         const unsigned* __restrict__ removed, unsigned* __restrict__ deg) {
  int e = blockIdx.x * blockDim.x + threadIdx.x;
  if (e >= E) return;
  if (!removed[e]) atomicAdd(&deg[rows[e]], 1u);
}

__global__ void k_dinv(const unsigned* __restrict__ deg, float* __restrict__ dinv, int n) {
  int i = blockIdx.x * blockDim.x + threadIdx.x;
  if (i < n) dinv[i] = 1.0f / sqrtf((float)deg[i] + 1e-12f);
}

__global__ void k_enc(const int* __restrict__ rows, const int* __restrict__ cols, int E,
                      const unsigned* __restrict__ removed, const float* __restrict__ dinv,
                      float* __restrict__ o_rows, float* __restrict__ o_cols,
                      float* __restrict__ o_enc) {
  int e = blockIdx.x * blockDim.x + threadIdx.x;
  if (e >= E) return;
  int r = rows[e], c = cols[e];
  o_rows[e] = (float)r;
  o_cols[e] = (float)c;
  o_enc[e]  = removed[e] ? 0.0f : dinv[r] * dinv[c];
}

// ---------------- mask augmentation + float view ----------------
__global__ void k_sample(unsigned* __restrict__ mask, int samp, int n) {
  int i = blockIdx.x * blockDim.x + threadIdx.x;
  if (i >= samp) return;
  uint32_t a, b;
  tf2x32(42u, 0xA5A5A5A5u, (uint32_t)i, 1u, a, b);
  mask[a % (unsigned)n] = 1u;
}

__global__ void k_mask2f(const unsigned* __restrict__ mask, float* __restrict__ xf, int n) {
  int i = blockIdx.x * blockDim.x + threadIdx.x;
  if (i < n) xf[i] = mask[i] ? 1.0f : 0.0f;
}

// ---------------- WMMA-based inclusive scan: 256 elems per wave ----------------
// y = inclusive_scan(x) within a 256-block; bsum[b] = block total.
// S = T16 * X (T lower-triangular ones) via v_wmma_f32_16x16x32_f16 (exact: ints <= 256),
// then a second WMMA adds the cross-column exclusive offsets.
__global__ __launch_bounds__(32) void k_scan256_wmma(const float* __restrict__ x,
                                                     float* __restrict__ y,
                                                     float* __restrict__ bsum, int n) {
  const int lane = threadIdx.x;
  const int base = blockIdx.x * 256;
  const bool hi = lane >= 16;
  const int nn = lane & 15;   // column N (B/C/D) and row M (A)

  // B: lanes 0-15 carry K=0..15 of column nn; lanes 16-31 carry K=16..31 (zeros here)
  v16h bm;
  for (int h = 0; h < 16; ++h) {
    float v = 0.0f;
    if (!hi) { int gi = base + nn * 16 + h; if (gi < n) v = x[gi]; }
    bm[h] = (_Float16)v;
  }
  // A: lower-triangular ones in first 16 K-columns. Contiguous K = 16*half + h (matches B).
  v16h am;
  for (int h = 0; h < 16; ++h) {
    int K = hi ? h + 16 : h;
    am[h] = (_Float16)((K < 16 && K <= nn) ? 1.0f : 0.0f);
  }
  v8f c = {};
  v8f s = __builtin_amdgcn_wmma_f32_16x16x32_f16(false, am, false, bm,
                                                 (short)0, c, false, false);
  // column sums: S[15][q] lives in VGPR7 lanes 16..31
  float s7 = s[7];
  float cs[16];
  for (int q = 0; q < 16; ++q) cs[q] = __shfl(s7, 16 + q, 32);
  // A2[m][k] = cs[k]; B2[k][n] = (k < n); D = A2*B2 + S  => off[n] + S[m][n]
  v16h a2, b2;
  for (int h = 0; h < 16; ++h) {
    int K = hi ? h + 16 : h;
    a2[h] = (_Float16)((K < 16) ? cs[K] : 0.0f);
    b2[h] = (_Float16)((!hi && h < nn) ? 1.0f : 0.0f);
  }
  v8f d = __builtin_amdgcn_wmma_f32_16x16x32_f16(false, a2, false, b2,
                                                 (short)0, s, false, false);
  for (int r = 0; r < 8; ++r) {
    int M = r + (hi ? 8 : 0);
    int gi = base + nn * 16 + M;
    if (gi < n) y[gi] = d[r];
  }
  if (lane == 31) bsum[blockIdx.x] = d[7];   // D[15][15] = block total
}

__global__ void k_scan_bsums(const float* __restrict__ bsum, float* __restrict__ boff, int nb) {
  if (blockIdx.x == 0 && threadIdx.x == 0) {
    float run = 0.0f;
    for (int i = 0; i < nb; ++i) { boff[i] = run; run += bsum[i]; }
    boff[nb] = run;   // total = tem_num
  }
}

__global__ void k_addoff(float* __restrict__ cum, const float* __restrict__ boff, int n) {
  int i = blockIdx.x * blockDim.x + threadIdx.x;
  if (i < n) cum[i] += boff[i >> 8];
}

// ---------------- key construction ----------------
__device__ __forceinline__ int lbound(const float* __restrict__ cum, int n, float target) {
  int lo = 0, hi = n;
  while (lo < hi) { int mid = (lo + hi) >> 1; if (cum[mid] < target) lo = mid + 1; else hi = mid; }
  return lo;
}

__global__ void k_buildkeys(const float* __restrict__ cum, const float* __restrict__ totalPtr,
                            int E, int n, unsigned long long* __restrict__ keys) {
  int e = blockIdx.x * blockDim.x + threadIdx.x;
  if (e >= E) return;
  __builtin_prefetch(&cum[(e * 37) % n], 0, 1);
  unsigned tem = (unsigned)(*totalPtr + 0.5f);
  if (tem == 0u) tem = 1u;
  uint32_t a, b;
  tf2x32(42u, 0x5EED5EEDu, (uint32_t)e, 2u, a, b);
  int tr = lbound(cum, n, (float)(a % tem) + 1.0f);
  int tc = lbound(cum, n, (float)(b % tem) + 1.0f);
  unsigned long long NN = (unsigned long long)NNODES;
  keys[e]     = (unsigned long long)tr * NN + (unsigned long long)tc + 1ull;
  keys[E + e] = (unsigned long long)tc * NN + (unsigned long long)tr + 1ull;
}

__global__ void k_selfloops(unsigned long long* __restrict__ keys, long long off, int n) {
  int i = blockIdx.x * blockDim.x + threadIdx.x;
  if (i < n)
    keys[off + i] = (unsigned long long)i * (unsigned long long)NNODES +
                    (unsigned long long)i + 1ull;
}

__global__ void k_keptkeys(const int* __restrict__ rows, const int* __restrict__ cols, int E,
                           const unsigned* __restrict__ removed,
                           unsigned long long* __restrict__ keys, long long off) {
  int e = blockIdx.x * blockDim.x + threadIdx.x;
  if (e >= E) return;
  unsigned long long k = 0ull;   // invalid -> 0 (sorts first, like -1 in ref)
  if (!removed[e])
    k = (unsigned long long)rows[e] * (unsigned long long)NNODES +
        (unsigned long long)cols[e] + 1ull;
  keys[off + e] = k;
}

// ---------------- LSD radix sort (8-bit digits, 5 passes) ----------------
__global__ __launch_bounds__(RS_THREADS)
void k_radix_hist(const unsigned long long* __restrict__ keys, long long M, int shift,
                  unsigned* __restrict__ hist, int nb) {
  __shared__ unsigned h[256];
  for (int d = threadIdx.x; d < 256; d += RS_THREADS) h[d] = 0u;
  __syncthreads();
  long long base = (long long)blockIdx.x * RS_TILE + (long long)threadIdx.x * RS_CPT;
  for (int j = 0; j < RS_CPT; ++j) {
    long long i = base + j;
    if (i < M) {
      if (i + RS_TILE < M) __builtin_prefetch(&keys[i + RS_TILE], 0, 1);
      unsigned d = (unsigned)((keys[i] >> shift) & 255ull);
      atomicAdd(&h[d], 1u);
    }
  }
  __syncthreads();
  for (int d = threadIdx.x; d < 256; d += RS_THREADS)
    hist[(long long)d * nb + blockIdx.x] = h[d];
}

__global__ __launch_bounds__(SCAN_THREADS)
void k_radix_scan(unsigned* __restrict__ hist, long long total) {
  __shared__ unsigned part[SCAN_THREADS];
  long long chunk = (total + SCAN_THREADS - 1) / SCAN_THREADS;
  long long b0 = (long long)threadIdx.x * chunk;
  long long b1 = b0 + chunk; if (b1 > total) b1 = total;
  unsigned s = 0u;
  for (long long i = b0; i < b1; ++i) s += hist[i];
  part[threadIdx.x] = s;
  __syncthreads();
  for (int off = 1; off < SCAN_THREADS; off <<= 1) {
    unsigned v = 0u;
    if ((int)threadIdx.x >= off) v = part[threadIdx.x - off];
    __syncthreads();
    part[threadIdx.x] += v;
    __syncthreads();
  }
  unsigned ex = (threadIdx.x == 0) ? 0u : part[threadIdx.x - 1];
  for (long long i = b0; i < b1; ++i) { unsigned v = hist[i]; hist[i] = ex; ex += v; }
}

__global__ __launch_bounds__(RS_THREADS)
void k_radix_scatter(const unsigned long long* __restrict__ src,
                     unsigned long long* __restrict__ dst,
                     long long M, int shift, const unsigned* __restrict__ hist, int nb) {
  __shared__ unsigned short cnt[RS_THREADS][256];   // 32 KB (big CDNA5 LDS)
  __shared__ unsigned gbase[256];
  const int t = threadIdx.x;
  for (int d = t; d < 256; d += RS_THREADS) gbase[d] = hist[(long long)d * nb + blockIdx.x];
  for (int d = 0; d < 256; ++d) cnt[t][d] = 0;
  __syncthreads();
  long long base = (long long)blockIdx.x * RS_TILE + (long long)t * RS_CPT;
  for (int j = 0; j < RS_CPT; ++j) {
    long long i = base + j;
    if (i < M) {
      unsigned d = (unsigned)((src[i] >> shift) & 255ull);
      cnt[t][d]++;
    }
  }
  __syncthreads();
  // exclusive scan across threads, per digit (each digit owned by one thread)
  for (int d = t; d < 256; d += RS_THREADS) {
    unsigned run = 0u;
    for (int tt = 0; tt < RS_THREADS; ++tt) {
      unsigned v = cnt[tt][d];
      cnt[tt][d] = (unsigned short)run;
      run += v;
    }
  }
  __syncthreads();
  for (int j = 0; j < RS_CPT; ++j) {
    long long i = base + j;
    if (i < M) {
      unsigned long long k = src[i];
      unsigned d = (unsigned)((k >> shift) & 255ull);
      unsigned pos = gbase[d] + cnt[t][d];
      cnt[t][d]++;
      dst[pos] = k;
    }
  }
}

// ---------------- finalize decoder adjacency ----------------
__global__ void k_finalize(const unsigned long long* __restrict__ keys, long long M,
                           float* __restrict__ drows, float* __restrict__ dcols,
                           float* __restrict__ dvals) {
  long long i = (long long)blockIdx.x * blockDim.x + threadIdx.x;
  if (i >= M) return;
  unsigned long long k = keys[i];
  bool first = (k > 0ull) && (i == 0 || keys[i - 1] != k);
  if (first) {
    unsigned long long h = k - 1ull;
    drows[i] = (float)(h / (unsigned long long)NNODES);
    dcols[i] = (float)(h % (unsigned long long)NNODES);
    dvals[i] = 1.0f;
  } else {
    drows[i] = 0.0f; dcols[i] = 0.0f; dvals[i] = 0.0f;
  }
}

// ---------------- host ----------------
extern "C" void kernel_launch(void* const* d_in, const int* in_sizes, int n_in,
                              void* d_out, int out_size, void* d_ws, size_t ws_size,
                              hipStream_t stream) {
  const int*   rows  = (const int*)d_in[0];
  const int*   cols  = (const int*)d_in[1];
  // d_in[2] (adj_vals) unused by the reference outputs we produce
  const int*   seeds = (const int*)d_in[3];
  const int E  = in_sizes[0];
  const int NS = in_sizes[3];
  const int N  = NNODES;
  const long long M = 3LL * E + N;                 // decoder edge-list length

  const int nb1 = (N + 255) / 256;                 // wmma-scan blocks
  const int nb  = (int)((M + RS_TILE - 1) / RS_TILE);

  // carve workspace
  size_t off = 0;
  auto carve = [&](size_t bytes) -> void* {
    void* p = (char*)d_ws + off;
    off += (bytes + 255) & ~(size_t)255;
    return p;
  };
  unsigned* seedA   = (unsigned*)carve((size_t)N * 4);
  unsigned* seedB   = (unsigned*)carve((size_t)N * 4);
  unsigned* mask    = (unsigned*)carve((size_t)N * 4);
  unsigned* removed = (unsigned*)carve((size_t)E * 4);
  unsigned* deg     = (unsigned*)carve((size_t)N * 4);
  float*    dinv    = (float*)carve((size_t)N * 4);
  float*    xf      = (float*)carve((size_t)N * 4);
  float*    cum     = (float*)carve((size_t)N * 4);
  float*    bsum    = (float*)carve((size_t)(nb1 + 2) * 4);
  float*    boff    = (float*)carve((size_t)(nb1 + 2) * 4);
  unsigned* hist    = (unsigned*)carve((size_t)256 * nb * 4);
  unsigned long long* keysA = (unsigned long long*)carve((size_t)M * 8);
  unsigned long long* keysB = (unsigned long long*)carve((size_t)M * 8);

  // output layout: rows | cols | enc_vals | dec_rows | dec_cols | dec_vals
  float* out     = (float*)d_out;
  float* o_rows  = out;
  float* o_cols  = out + (size_t)E;
  float* o_enc   = out + 2 * (size_t)E;
  float* o_drows = out + 3 * (size_t)E;
  float* o_dcols = o_drows + (size_t)M;
  float* o_dvals = o_dcols + (size_t)M;

  hipMemsetAsync(seedA,   0, (size_t)N * 4, stream);
  hipMemsetAsync(seedB,   0, (size_t)N * 4, stream);
  hipMemsetAsync(mask,    0, (size_t)N * 4, stream);
  hipMemsetAsync(removed, 0, (size_t)E * 4, stream);
  hipMemsetAsync(deg,     0, (size_t)N * 4, stream);

  const int T = 256;
  const int gE = (E + T - 1) / T;
  const int gN = (N + T - 1) / T;

  k_scatter_seeds<<<(NS + T - 1) / T, T, 0, stream>>>(seeds, NS, seedA, mask);
  k_bfs0<<<gE, T, 0, stream>>>(rows, cols, E, seedA, removed, seedB, mask);
  k_bfs1<<<gE, T, 0, stream>>>(rows, cols, E, seedB, removed);

  k_degree<<<gE, T, 0, stream>>>(rows, E, removed, deg);
  k_dinv<<<gN, T, 0, stream>>>(deg, dinv, N);
  k_enc<<<gE, T, 0, stream>>>(rows, cols, E, removed, dinv, o_rows, o_cols, o_enc);

  const int samp = (int)((long long)N * 2 / 10);   // KEEP_RATE = 0.2
  k_sample<<<(samp + T - 1) / T, T, 0, stream>>>(mask, samp, N);
  k_mask2f<<<gN, T, 0, stream>>>(mask, xf, N);

  // WMMA inclusive scan of mask -> cum, total -> boff[nb1]
  k_scan256_wmma<<<nb1, 32, 0, stream>>>(xf, cum, bsum, N);
  k_scan_bsums<<<1, 32, 0, stream>>>(bsum, boff, nb1);
  k_addoff<<<gN, T, 0, stream>>>(cum, boff, N);

  // build 12.15M hash keys (bias +1; 0 = invalid, sorts first like -1)
  k_buildkeys<<<gE, T, 0, stream>>>(cum, &boff[nb1], E, N, keysA);
  k_selfloops<<<gN, T, 0, stream>>>(keysA, 2LL * E, N);
  k_keptkeys<<<gE, T, 0, stream>>>(rows, cols, E, removed, keysA, 2LL * E + N);

  // LSD radix sort: 5 x 8-bit passes (keys < 2^35)
  unsigned long long* src = keysA;
  unsigned long long* dst = keysB;
  for (int pass = 0; pass < 5; ++pass) {
    int shift = pass * 8;
    k_radix_hist<<<nb, RS_THREADS, 0, stream>>>(src, M, shift, hist, nb);
    k_radix_scan<<<1, SCAN_THREADS, 0, stream>>>(hist, 256LL * nb);
    k_radix_scatter<<<nb, RS_THREADS, 0, stream>>>(src, dst, M, shift, hist, nb);
    unsigned long long* tmp = src; src = dst; dst = tmp;
  }
  // after 5 passes, sorted data is in `src` (keysB)

  k_finalize<<<(int)((M + T - 1) / T), T, 0, stream>>>(src, M, o_drows, o_dcols, o_dvals);
}
